// PointerNet_26980984554045
// MI455X (gfx1250) — compile-verified, matrix operand order checked by reading, not run on hardware
//
#include <hip/hip_runtime.h>
#include <cstdint>

// ---------------------------------------------------------------------------
// PointerNet forward for MI455X (gfx1250, wave32, WMMA).
// B=2048, S=128, H=128, E=128, 4H=512.  All GEMMs run on
// v_wmma_f32_16x16x32_f16 (fp16 in / fp32 accumulate), K hardcoded to 128,
// A fragments register-cached and reused across 4 N-tiles per wave.
// ---------------------------------------------------------------------------

#define BB 2048
#define SS 128
#define HH 128
#define EE 128
#define H4 512
#define KK 128
#define C_SCALE 10.0f

typedef __attribute__((ext_vector_type(16))) _Float16 v16h;
typedef __attribute__((ext_vector_type(8)))  _Float16 v8h;
typedef __attribute__((ext_vector_type(8)))  float    v8f;

// ---------------- WMMA fragment load (16-bit A-matrix 16x32 layout) --------
// lane 0-15: row = lane, k-base 0 ; lane 16-31: row = lane-16, k-base 8.
// frag halves [0..7]  = k (kb .. kb+7)        -> one 16B load
// frag halves [8..15] = k (kb+16 .. kb+23)    -> one 16B load
__device__ __forceinline__ v16h load_frag(const _Float16* p) {
  v8h lo = *(const v8h*)(p);
  v8h hi = *(const v8h*)(p + 16);
  v16h f;
#pragma unroll
  for (int i = 0; i < 8; ++i) { f[i] = lo[i]; f[i + 8] = hi[i]; }
  return f;
}

__device__ __forceinline__ v8f wmma_f16(v16h a, v16h b, v8f c) {
  return __builtin_amdgcn_wmma_f32_16x16x32_f16(false, a, false, b,
                                                (short)0, c, false, false);
}

// Out[m,n] = sum_k A1[m,k]*W1[n,k] (+ sum_k A2[m,k]*W2[n,k]) + bias[n]
// A row strides lda1/lda2 (elements); W row-major [N,K=128].
// Each wave: one 16-row M block x NT consecutive 16-col N tiles.
// A fragments (all 4 K-chunks) cached in VGPRs, reused across the NT tiles.
// grid = (N/16/NT, M/16), block = 32; exact fit -> EXEC all-ones.
template <int NT, bool DUAL, bool STOREHALF>
__global__ void k_gemm_wmma(const _Float16* __restrict__ A1, int lda1,
                            const _Float16* __restrict__ W1,
                            const _Float16* __restrict__ A2, int lda2,
                            const _Float16* __restrict__ W2,
                            const float* __restrict__ bias,
                            void* __restrict__ Out, int outRS, int outCS) {
  const int lane = threadIdx.x;                  // 0..31
  const int tm = blockIdx.y, tn0 = blockIdx.x * NT;
  const int r16 = lane & 15;
  const int kb  = (lane >> 4) * 8;

  v16h a1f[4], a2f[4];
  {
    const _Float16* a = A1 + (size_t)(tm * 16 + r16) * lda1 + kb;
#pragma unroll
    for (int c = 0; c < 4; ++c) a1f[c] = load_frag(a + c * 32);
  }
  if (DUAL) {
    const _Float16* a = A2 + (size_t)(tm * 16 + r16) * lda2 + kb;
#pragma unroll
    for (int c = 0; c < 4; ++c) a2f[c] = load_frag(a + c * 32);
  }

#pragma unroll
  for (int j = 0; j < NT; ++j) {
    const int tn = tn0 + j;
    const _Float16* w1 = W1 + (size_t)(tn * 16 + r16) * KK + kb;
    v8f acc = {};
#pragma unroll
    for (int c = 0; c < 4; ++c)
      acc = wmma_f16(a1f[c], load_frag(w1 + c * 32), acc);
    if (DUAL) {
      const _Float16* w2 = W2 + (size_t)(tn * 16 + r16) * KK + kb;
#pragma unroll
      for (int c = 0; c < 4; ++c)
        acc = wmma_f16(a2f[c], load_frag(w2 + c * 32), acc);
    }
    // C/D layout: lanes 0-15 -> M rows 0..7, lanes 16-31 -> 8..15; N = lane&15
    const int n  = tn * 16 + (lane & 15);
    const int mb = tm * 16 + (lane >> 4) * 8;
    const float bv = bias ? bias[n] : 0.0f;
    if (STOREHALF) {
      _Float16* o = (_Float16*)Out;
#pragma unroll
      for (int r = 0; r < 8; ++r)
        o[(size_t)(mb + r) * outRS + (size_t)n * outCS] = (_Float16)(acc[r] + bv);
    } else {
      float* o = (float*)Out;
#pragma unroll
      for (int r = 0; r < 8; ++r)
        o[(size_t)(mb + r) * outRS + (size_t)n * outCS] = acc[r] + bv;
    }
  }
}

// ---------------- small elementwise / setup kernels ------------------------
__global__ void k_f2h(const float* __restrict__ s, _Float16* __restrict__ d, int n) {
  int i = blockIdx.x * blockDim.x + threadIdx.x;
  if (i < n) d[i] = (_Float16)s[i];
}

__global__ void k_bias2(const float* __restrict__ a, const float* __restrict__ b,
                        float* __restrict__ d, int n) {
  int i = blockIdx.x * blockDim.x + threadIdx.x;
  if (i < n) d[i] = a[i] + b[i];
}

// embedded[b,s,e] = in[b,0,s]*W[0,e] + in[b,1,s]*W[1,e]   (fp16 out)
__global__ void k_embed(const float* __restrict__ in, const float* __restrict__ W,
                        _Float16* __restrict__ out) {
  int bs = blockIdx.x;
  int b = bs / SS, s = bs % SS, e = threadIdx.x;
  float x0 = in[((size_t)b * 2 + 0) * SS + s];
  float x1 = in[((size_t)b * 2 + 1) * SS + s];
  out[(size_t)bs * EE + e] = (_Float16)(x0 * W[e] + x1 * W[EE + e]);
}

__global__ void k_init(float* __restrict__ h32, float* __restrict__ c32,
                       _Float16* __restrict__ h16, unsigned char* __restrict__ mask,
                       _Float16* __restrict__ dec_in, const float* __restrict__ start) {
  int i = blockIdx.x * blockDim.x + threadIdx.x;   // covers B*H == B*S == B*E
  if (i < BB * HH) { h32[i] = 0.f; c32[i] = 0.f; h16[i] = (_Float16)0.f; }
  if (i < BB * SS) mask[i] = 0;
  if (i < BB * EE) dec_in[i] = (_Float16)start[i % EE];
}

__device__ __forceinline__ float sigm(float x) { return 1.0f / (1.0f + __expf(-x)); }

__global__ void k_lstm_cell(const float* __restrict__ gates, float* __restrict__ h32,
                            float* __restrict__ c32, _Float16* __restrict__ h16) {
  int b = blockIdx.x, j = threadIdx.x;
  const float* g = gates + (size_t)b * H4;
  float gi = g[j], gf = g[HH + j], gg = g[2 * HH + j], go = g[3 * HH + j];
  float c = c32[(size_t)b * HH + j];
  c = sigm(gf) * c + sigm(gi) * tanhf(gg);
  float h = sigm(go) * tanhf(c);
  c32[(size_t)b * HH + j] = c;
  h32[(size_t)b * HH + j] = h;
  h16[(size_t)b * HH + j] = (_Float16)h;
}

// score[b,s] = sum_h V[h]*tanh(q[b,h] + ref[b,h,s]); optional 10*tanh; mask->-inf
__global__ void k_att_score(const float* __restrict__ q, const _Float16* __restrict__ ref,
                            const float* __restrict__ V,
                            const unsigned char* __restrict__ mask,
                            float* __restrict__ out, int ctanh) {
  int b = blockIdx.x, s = threadIdx.x;
  const _Float16* r = ref + (size_t)b * HH * SS + s;
  const float* qb = q + (size_t)b * HH;
  float acc = 0.f;
#pragma unroll 4
  for (int h = 0; h < HH; ++h)
    acc += V[h] * tanhf(qb[h] + (float)r[(size_t)h * SS]);
  if (ctanh) acc = C_SCALE * tanhf(acc);
  if (mask[(size_t)b * SS + s]) acc = -__builtin_inff();
  out[(size_t)b * SS + s] = acc;
}

// softmax(gl[b,:]) ; q2[b,h] = sum_s ref[b,h,s]*w[s]  (fp16 out)
__global__ void k_glimpse(const float* __restrict__ gl, const _Float16* __restrict__ ref,
                          _Float16* __restrict__ q2) {
  __shared__ float w[SS];
  __shared__ float red[SS];
  int b = blockIdx.x, t = threadIdx.x;
  float v = gl[(size_t)b * SS + t];
  red[t] = v; __syncthreads();
  for (int off = SS / 2; off; off >>= 1) {
    if (t < off) red[t] = fmaxf(red[t], red[t + off]);
    __syncthreads();
  }
  float m = red[0]; __syncthreads();
  float e = __expf(v - m);
  red[t] = e; __syncthreads();
  for (int off = SS / 2; off; off >>= 1) {
    if (t < off) red[t] += red[t + off];
    __syncthreads();
  }
  float sum = red[0];
  w[t] = e / sum; __syncthreads();
  const _Float16* r = ref + ((size_t)b * HH + t) * SS;   // t acts as h
  float acc = 0.f;
#pragma unroll 4
  for (int s = 0; s < SS; ++s) acc += (float)r[s] * w[s];
  q2[(size_t)b * HH + t] = (_Float16)acc;
}

__device__ __forceinline__ float hash01(uint32_t x) {
  x ^= x >> 16; x *= 0x7feb352du; x ^= x >> 15; x *= 0x846ca68bu; x ^= x >> 16;
  return ((x >> 8) + 0.5f) * (1.0f / 16777216.0f);
}

// softmax -> probs_out; Gumbel-max sample -> idx; mask update; dec_in gather
__global__ void k_sample(const float* __restrict__ pl, const _Float16* __restrict__ emb,
                         unsigned char* __restrict__ mask, _Float16* __restrict__ dec_in,
                         float* __restrict__ probs_out, int* __restrict__ idx_out, int t) {
  __shared__ float red[SS];
  __shared__ int   redi[SS];
  __shared__ int   sidx;
  int b = blockIdx.x, s = threadIdx.x;
  float v = pl[(size_t)b * SS + s];
  red[s] = v; __syncthreads();
  for (int off = SS / 2; off; off >>= 1) {
    if (s < off) red[s] = fmaxf(red[s], red[s + off]);
    __syncthreads();
  }
  float m = red[0]; __syncthreads();
  float e = __expf(v - m);
  red[s] = e; __syncthreads();
  for (int off = SS / 2; off; off >>= 1) {
    if (s < off) red[s] += red[s + off];
    __syncthreads();
  }
  float sum = red[0];
  probs_out[(size_t)b * SS + s] = e / sum;
  __syncthreads();
  // deterministic Gumbel-max categorical sample
  uint32_t seed = ((uint32_t)t * 2654435761u) ^ ((uint32_t)b * 0x9E3779B9u)
                ^ ((uint32_t)s * 0x85EBCA6Bu) ^ 0xC0FFEE42u;
  float g = v - __logf(-__logf(hash01(seed)));
  red[s] = g; redi[s] = s; __syncthreads();
  for (int off = SS / 2; off; off >>= 1) {
    if (s < off && red[s + off] > red[s]) { red[s] = red[s + off]; redi[s] = redi[s + off]; }
    __syncthreads();
  }
  if (s == 0) {
    int id = redi[0];
    sidx = id;
    idx_out[b] = id;
    mask[(size_t)b * SS + id] = 1;
  }
  __syncthreads();
  int id = sidx;                                   // s acts as e (E==S==128)
  dec_in[(size_t)b * EE + s] = emb[((size_t)b * SS + id) * EE + s];
}

// ---------------------------------------------------------------------------
extern "C" void kernel_launch(void* const* d_in, const int* in_sizes, int n_in,
                              void* d_out, int out_size, void* d_ws, size_t ws_size,
                              hipStream_t stream) {
  (void)in_sizes; (void)n_in; (void)out_size; (void)ws_size;

  const float* in_pts   = (const float*)d_in[0];
  const float* emb_W    = (const float*)d_in[1];
  const float* enc_Wih  = (const float*)d_in[2];
  const float* enc_Whh  = (const float*)d_in[3];
  const float* enc_bih  = (const float*)d_in[4];
  const float* enc_bhh  = (const float*)d_in[5];
  const float* dec_Wih  = (const float*)d_in[6];
  const float* dec_Whh  = (const float*)d_in[7];
  const float* dec_bih  = (const float*)d_in[8];
  const float* dec_bhh  = (const float*)d_in[9];
  const float* g_Wq     = (const float*)d_in[10];
  const float* g_bq     = (const float*)d_in[11];
  const float* g_Wref   = (const float*)d_in[12];
  const float* g_bref   = (const float*)d_in[13];
  const float* g_V      = (const float*)d_in[14];
  const float* p_Wq     = (const float*)d_in[15];
  const float* p_bq     = (const float*)d_in[16];
  const float* p_Wref   = (const float*)d_in[17];
  const float* p_bref   = (const float*)d_in[18];
  const float* p_V      = (const float*)d_in[19];
  const float* start    = (const float*)d_in[20];

  float* probs_base = (float*)d_out;                               // [S,B,S]
  int*   idx_base   = (int*)d_out + (size_t)SS * BB * SS;          // [S,B]

  // ---- workspace layout (256B aligned) ----
  char* ws = (char*)d_ws;
  size_t off = 0;
  auto alloc = [&](size_t bytes) -> char* {
    char* p = ws + off;
    off = (off + bytes + 255) & ~(size_t)255;
    return p;
  };
  _Float16* emb_h   = (_Float16*)alloc((size_t)BB * SS * EE * 2);  // 64 MB
  _Float16* g_ref   = (_Float16*)alloc((size_t)BB * HH * SS * 2);  // 64 MB
  _Float16* p_ref   = (_Float16*)alloc((size_t)BB * HH * SS * 2);  // 64 MB
  _Float16* h16     = (_Float16*)alloc((size_t)BB * HH * 2);
  _Float16* dec_in  = (_Float16*)alloc((size_t)BB * EE * 2);
  _Float16* q2_16   = (_Float16*)alloc((size_t)BB * HH * 2);
  float*    h32     = (float*)alloc((size_t)BB * HH * 4);
  float*    c32     = (float*)alloc((size_t)BB * HH * 4);
  float*    gates   = (float*)alloc((size_t)BB * H4 * 4);
  float*    q32     = (float*)alloc((size_t)BB * HH * 4);
  float*    gl      = (float*)alloc((size_t)BB * SS * 4);
  float*    pl      = (float*)alloc((size_t)BB * SS * 4);
  unsigned char* mask = (unsigned char*)alloc((size_t)BB * SS);
  _Float16* eWih_h  = (_Float16*)alloc((size_t)H4 * EE * 2);
  _Float16* eWhh_h  = (_Float16*)alloc((size_t)H4 * HH * 2);
  _Float16* dWih_h  = (_Float16*)alloc((size_t)H4 * EE * 2);
  _Float16* dWhh_h  = (_Float16*)alloc((size_t)H4 * HH * 2);
  _Float16* gWq_h   = (_Float16*)alloc((size_t)HH * HH * 2);
  _Float16* gWr_h   = (_Float16*)alloc((size_t)HH * HH * 2);
  _Float16* pWq_h   = (_Float16*)alloc((size_t)HH * HH * 2);
  _Float16* pWr_h   = (_Float16*)alloc((size_t)HH * HH * 2);
  float*    enc_b   = (float*)alloc(H4 * 4);
  float*    dec_b   = (float*)alloc(H4 * 4);

  // ---- one-time (per call) prep: weight conversions, embed, state init ----
  auto f2h = [&](const float* s, _Float16* d, int n) {
    k_f2h<<<(n + 255) / 256, 256, 0, stream>>>(s, d, n);
  };
  f2h(enc_Wih, eWih_h, H4 * EE);  f2h(enc_Whh, eWhh_h, H4 * HH);
  f2h(dec_Wih, dWih_h, H4 * EE);  f2h(dec_Whh, dWhh_h, H4 * HH);
  f2h(g_Wq, gWq_h, HH * HH);      f2h(g_Wref, gWr_h, HH * HH);
  f2h(p_Wq, pWq_h, HH * HH);      f2h(p_Wref, pWr_h, HH * HH);
  k_bias2<<<(H4 + 255) / 256, 256, 0, stream>>>(enc_bih, enc_bhh, enc_b, H4);
  k_bias2<<<(H4 + 255) / 256, 256, 0, stream>>>(dec_bih, dec_bhh, dec_b, H4);
  k_embed<<<BB * SS, EE, 0, stream>>>(in_pts, emb_W, emb_h);
  k_init<<<(BB * SS + 255) / 256, 256, 0, stream>>>(h32, c32, h16, mask, dec_in, start);

  const dim3 gGates(H4 / 64, BB / 16);   // NT=4, N=512
  const dim3 gProj (HH / 64, BB / 16);   // NT=4, N=128

  // ---- encoder: 128 sequential steps ----
  for (int t = 0; t < SS; ++t) {
    // gates = x_t@Wih^T + h@Whh^T + (bih+bhh)
    k_gemm_wmma<4, true, false><<<gGates, 32, 0, stream>>>(
        emb_h + (size_t)t * EE, SS * EE, eWih_h, h16, HH, eWhh_h,
        enc_b, gates, H4, 1);
    k_lstm_cell<<<BB, HH, 0, stream>>>(gates, h32, c32, h16);
    // ref projections for this timestep (strided fp16 store into [B,H,S])
    k_gemm_wmma<4, false, true><<<gProj, 32, 0, stream>>>(
        h16, HH, gWr_h, nullptr, 0, nullptr, g_bref,
        (void*)(g_ref + t), HH * SS, SS);
    k_gemm_wmma<4, false, true><<<gProj, 32, 0, stream>>>(
        h16, HH, pWr_h, nullptr, 0, nullptr, p_bref,
        (void*)(p_ref + t), HH * SS, SS);
  }

  // ---- decoder: 128 sequential steps ----
  for (int t = 0; t < SS; ++t) {
    k_gemm_wmma<4, true, false><<<gGates, 32, 0, stream>>>(
        dec_in, EE, dWih_h, h16, HH, dWhh_h, dec_b, gates, H4, 1);
    k_lstm_cell<<<BB, HH, 0, stream>>>(gates, h32, c32, h16);
    // glimpse
    k_gemm_wmma<4, false, false><<<gProj, 32, 0, stream>>>(
        h16, HH, gWq_h, nullptr, 0, nullptr, g_bq, q32, HH, 1);
    k_att_score<<<BB, SS, 0, stream>>>(q32, g_ref, g_V, mask, gl, 0);
    k_glimpse<<<BB, SS, 0, stream>>>(gl, g_ref, q2_16);
    // pointer
    k_gemm_wmma<4, false, false><<<gProj, 32, 0, stream>>>(
        q2_16, HH, pWq_h, nullptr, 0, nullptr, p_bq, q32, HH, 1);
    k_att_score<<<BB, SS, 0, stream>>>(q32, p_ref, p_V, mask, pl, 1);
    k_sample<<<BB, SS, 0, stream>>>(pl, emb_h, mask, dec_in,
                                    probs_base + (size_t)t * BB * SS,
                                    idx_base + (size_t)t * BB, t);
  }
}
